// MVSNet_29171417875210
// MI455X (gfx1250) — matrix operand order repaired and checked by reading, not compiled, not run on hardware
//
#include <hip/hip_runtime.h>
#include <hip/hip_bf16.h>

typedef __attribute__((ext_vector_type(16))) _Float16 v16h;
typedef __attribute__((ext_vector_type(8)))  float    v8f;

#define D0 48
#define H0 112
#define W0 144
#define HW0 (H0*W0)        /* 16128  */
#define NP0 (D0*H0*W0)     /* 774144 */
#define D1 24
#define H1 56
#define W1 72
#define NP1 (D1*H1*W1)     /* 96768  */
#define BCH 16

// ---------------------------------------------------------------- projection
__global__ void mvs_proj_kernel(const float* __restrict__ refP,
                                const float* __restrict__ srcP,
                                float* __restrict__ out) {
  if (threadIdx.x != 0 || blockIdx.x != 0) return;
  float a[4][8];
  for (int i = 0; i < 4; ++i)
    for (int j = 0; j < 4; ++j) { a[i][j] = refP[i*4+j]; a[i][4+j] = (i==j) ? 1.f : 0.f; }
  for (int c = 0; c < 4; ++c) {
    int piv = c;
    for (int r = c+1; r < 4; ++r) if (fabsf(a[r][c]) > fabsf(a[piv][c])) piv = r;
    if (piv != c) for (int j = 0; j < 8; ++j) { float t=a[c][j]; a[c][j]=a[piv][j]; a[piv][j]=t; }
    float ip = 1.f / a[c][c];
    for (int j = 0; j < 8; ++j) a[c][j] *= ip;
    for (int r = 0; r < 4; ++r) if (r != c) {
      float f = a[r][c];
      for (int j = 0; j < 8; ++j) a[r][j] -= f * a[c][j];
    }
  }
  for (int v = 0; v < 2; ++v) {
    const float* S = srcP + v*16;
    for (int i = 0; i < 3; ++i)
      for (int j = 0; j < 4; ++j) {
        float s = 0.f;
        for (int t = 0; t < 4; ++t) s += S[i*4+t] * a[t][4+j];
        if (j < 3) out[v*12 + i*3 + j] = s;
        else       out[v*12 + 9 + i]   = s;
      }
  }
}

// ---------------------------------------------------------- warped cost volume
__global__ void mvs_cost_volume(const float* __restrict__ ref,
                                const float* __restrict__ src,
                                const float* __restrict__ dep,
                                const float* __restrict__ proj,
                                _Float16* __restrict__ cv) {
  int idx = blockIdx.x * blockDim.x + threadIdx.x;
  if (idx >= NP0) return;
  int x = idx % W0; int t = idx / W0; int y = t % H0;
  float depth = dep[idx];
  float s[BCH], q[BCH];
  #pragma unroll
  for (int c = 0; c < BCH; ++c) { float r = ref[c*HW0 + y*W0 + x]; s[c] = r; q[c] = r*r; }
  for (int v = 0; v < 2; ++v) {
    const float* R = proj + v*12;
    float rx = R[0]*x + R[1]*y + R[2];
    float ry = R[3]*x + R[4]*y + R[5];
    float rz = R[6]*x + R[7]*y + R[8];
    float X = rx*depth + R[9];
    float Y = ry*depth + R[10];
    float Z = rz*depth + R[11];
    if (fabsf(Z) < 1e-6f) Z = 1e-6f;
    float px = X / Z, py = Y / Z;
    float fx0 = floorf(px), fy0 = floorf(py);
    float wx = px - fx0, wy = py - fy0;
    float wgt[4] = {(1.f-wx)*(1.f-wy), wx*(1.f-wy), (1.f-wx)*wy, wx*wy};
    float cx[4]  = {fx0, fx0+1.f, fx0, fx0+1.f};
    float cy[4]  = {fy0, fy0, fy0+1.f, fy0+1.f};
    float wp[BCH];
    #pragma unroll
    for (int c = 0; c < BCH; ++c) wp[c] = 0.f;
    for (int k = 0; k < 4; ++k) {
      bool valid = (cx[k] >= 0.f) && (cx[k] <= (float)(W0-1)) &&
                   (cy[k] >= 0.f) && (cy[k] <= (float)(H0-1));
      if (!valid) continue;
      int xi = (int)cx[k]; int yi = (int)cy[k];
      int p2 = yi*W0 + xi;
      float wk = wgt[k];
      #pragma unroll
      for (int c = 0; c < BCH; ++c) wp[c] += src[(v*BCH + c)*HW0 + p2] * wk;
    }
    #pragma unroll
    for (int c = 0; c < BCH; ++c) { s[c] += wp[c]; q[c] += wp[c]*wp[c]; }
  }
  const float inv3 = 1.f / 3.f;
  #pragma unroll
  for (int c = 0; c < BCH; ++c) {
    float m = s[c]*inv3;
    cv[(size_t)c*NP0 + idx] = (_Float16)(q[c]*inv3 - m*m);
  }
}

// ---------------------------------------------- implicit-GEMM 3D conv via WMMA
// K padded to 32 per input channel: one WMMA chunk == one input channel.
// A: unconditional saddr+voffset u16 loads + branchless cndmask zeroing.
// B: weights staged in LDS padded [cout16][ci][tap32] -> one aligned v16h read.
__global__ __launch_bounds__(128) void mvs_conv3d_wmma(
    const _Float16* __restrict__ in, const float* __restrict__ wt,
    float* __restrict__ out,
    int Cin, int Din, int Hin, int Win,
    int Dout, int Hout, int Wout, int stride)
{
  __shared__ _Float16 wlds[16 * 64 * 32];       // padded taps, <= 64 KB
  const int Ktot     = Cin * 27;
  const int nbase    = blockIdx.y * 16;
  const int npix_in  = Din * Hin * Win;
  const int npix_out = Dout * Hout * Wout;

  // stage this Cout-block's weights (f32 -> f16), taps padded 27 -> 32 with zeros
  for (int i = threadIdx.x; i < 16 * Cin * 32; i += 128) {
    int tap = i & 31; int nc = i >> 5; int ci = nc % Cin; int n = nc / Cin;
    float v = (tap < 27) ? wt[(size_t)(nbase + n) * Ktot + ci*27 + tap] : 0.f;
    wlds[i] = (_Float16)v;
  }
  __syncthreads();

  const int lane  = threadIdx.x & 31;
  const int wave  = threadIdx.x >> 5;
  const int half  = lane >> 4;
  const int ln    = lane & 15;
  const int mwave = (blockIdx.x * 4 + wave) * 32;   // two 16-row tiles per wave
  if (mwave >= npix_out) return;                    // wave-uniform

  const int m0 = mwave + ln;
  const int m1 = m0 + 16;
  int xo0 = m0 % Wout, tt0 = m0 / Wout, yo0 = tt0 % Hout, zo0 = tt0 / Hout;
  int xo1 = m1 % Wout, tt1 = m1 / Wout, yo1 = tt1 % Hout, zo1 = tt1 / Hout;
  const int zb0 = zo0*stride - 1, yb0 = yo0*stride - 1, xb0 = xo0*stride - 1;
  const int zb1 = zo1*stride - 1, yb1 = yo1*stride - 1, xb1 = xo1*stride - 1;

  // precompute A gather offsets + validity (contiguous k = 16*half + i, padded taps)
  int aofs0[16], aofs1[16];
  unsigned amask0 = 0, amask1 = 0;
  #pragma unroll
  for (int i = 0; i < 16; ++i) {
    int r  = half*16 + i;                           // tap id 0..31 (>=27 -> pad)
    int dz = r / 9; int rem = r - dz*9; int dy = rem / 3; int dx = rem - dy*3;
    {
      int zi = zb0 + dz, yi = yb0 + dy, xi = xb0 + dx;
      bool ok = (r < 27) && (unsigned)zi < (unsigned)Din &&
                (unsigned)yi < (unsigned)Hin && (unsigned)xi < (unsigned)Win;
      aofs0[i] = ok ? ((zi*Hin + yi)*Win + xi) : 0;
      amask0  |= ok ? (1u << i) : 0u;
    }
    {
      int zi = zb1 + dz, yi = yb1 + dy, xi = xb1 + dx;
      bool ok = (r < 27) && (unsigned)zi < (unsigned)Din &&
                (unsigned)yi < (unsigned)Hin && (unsigned)xi < (unsigned)Win;
      aofs1[i] = ok ? ((zi*Hin + yi)*Win + xi) : 0;
      amask1  |= ok ? (1u << i) : 0u;
    }
  }

  const _Float16* wrow = &wlds[(size_t)ln * Cin * 32];   // this lane's cout row
  v8f acc0 = {}, acc1 = {};
  const _Float16* base = in;
  for (int ci = 0; ci < Cin; ++ci, base += npix_in) {
    __builtin_prefetch(base + npix_in, 0, 1);            // next channel plane
    v16h A0, A1;
    #pragma unroll
    for (int i = 0; i < 16; ++i) {                       // loads are UNconditional
      _Float16 v = base[aofs0[i]];
      A0[i] = ((amask0 >> i) & 1) ? v : (_Float16)0.f;   // v_cndmask, no branch
    }
    #pragma unroll
    for (int i = 0; i < 16; ++i) {
      _Float16 v = base[aofs1[i]];
      A1[i] = ((amask1 >> i) & 1) ? v : (_Float16)0.f;
    }
    v16h Bm = *(const v16h*)&wrow[ci*32 + half*16];      // 32B aligned LDS read
    acc0 = __builtin_amdgcn_wmma_f32_16x16x32_f16(false, A0, false, Bm,
                                                  (short)0, acc0, false, false);
    acc1 = __builtin_amdgcn_wmma_f32_16x16x32_f16(false, A1, false, Bm,
                                                  (short)0, acc1, false, false);
  }

  // C/D layout: VGPR r, lane l -> (M = r + 8*(l>>4), N = l&15)
  const int n = nbase + ln;
  #pragma unroll
  for (int r = 0; r < 8; ++r) {
    int mr = mwave + r + 8*half;
    out[(size_t)n * npix_out + mr]      = acc0[r];
    out[(size_t)n * npix_out + mr + 16] = acc1[r];
  }
}

// -------------------------------------------------------------- batchnorm path
__global__ void mvs_bn_stats(const float* __restrict__ x, float* __restrict__ stats,
                             int npix) {
  __shared__ float ssum[256], ssq[256];
  int c = blockIdx.x;
  const float* p = x + (size_t)c * npix;
  float s = 0.f, q = 0.f;
  for (int i = threadIdx.x; i < npix; i += 256) { float v = p[i]; s += v; q += v*v; }
  ssum[threadIdx.x] = s; ssq[threadIdx.x] = q;
  __syncthreads();
  for (int off = 128; off > 0; off >>= 1) {
    if ((int)threadIdx.x < off) {
      ssum[threadIdx.x] += ssum[threadIdx.x + off];
      ssq[threadIdx.x]  += ssq[threadIdx.x + off];
    }
    __syncthreads();
  }
  if (threadIdx.x == 0) {
    float mean = ssum[0] / npix;
    float var  = ssq[0] / npix - mean*mean;
    stats[2*c]   = mean;
    stats[2*c+1] = rsqrtf(var + 1e-5f);
  }
}

// bn+relu (+optional f16 skip add), writes f16 activation; channel = blockIdx.y
__global__ void mvs_bn_relu_add(const float* __restrict__ x, const float* __restrict__ stats,
                                const float* __restrict__ g, const float* __restrict__ b,
                                const _Float16* __restrict__ skip, _Float16* __restrict__ y,
                                int npix) {
  int i = blockIdx.x*blockDim.x + threadIdx.x;
  if (i >= npix) return;
  int c = blockIdx.y;
  size_t idx = (size_t)c*npix + i;
  float v = (x[idx] - stats[2*c]) * stats[2*c+1] * g[c] + b[c];
  v = v > 0.f ? v : 0.f;
  if (skip) v += (float)skip[idx];
  y[idx] = (_Float16)v;
}

// --------------------------------------------- transposed conv (32->16, x2 up)
__global__ void mvs_convT_s2(const _Float16* __restrict__ in, const float* __restrict__ wt,
                             float* __restrict__ out) {
  int idx = blockIdx.x*blockDim.x + threadIdx.x;
  const int total = 16 * NP0;
  if (idx >= total) return;
  int m = idx % NP0;
  int o = idx / NP0;
  int x = m % W0; int t = m / W0; int y = t % H0; int z = t / H0;
  float acc = 0.f;
  for (int kz = 0; kz < 3; ++kz) {
    int qz = z - 1 + kz;
    if (qz < 0 || qz > 2*D1-2 || (qz & 1)) continue;
    int iz = qz >> 1;
    for (int ky = 0; ky < 3; ++ky) {
      int qy = y - 1 + ky;
      if (qy < 0 || qy > 2*H1-2 || (qy & 1)) continue;
      int iy = qy >> 1;
      for (int kx = 0; kx < 3; ++kx) {
        int qx = x - 1 + kx;
        if (qx < 0 || qx > 2*W1-2 || (qx & 1)) continue;
        int ix = qx >> 1;
        int sp = (iz*H1 + iy)*W1 + ix;
        const float* wp = wt + (size_t)(o*32)*27 + (kz*9 + ky*3 + kx);
        for (int ci = 0; ci < 32; ++ci)
          acc += (float)in[(size_t)ci*NP1 + sp] * wp[(size_t)ci*27];
      }
    }
  }
  out[(size_t)o*NP0 + m] = acc;
}

// ------------------------------------------------------- final 16->1 prob conv
__global__ void mvs_prob_conv(const _Float16* __restrict__ in, const float* __restrict__ wt,
                              const float* __restrict__ bias, float* __restrict__ out) {
  int idx = blockIdx.x*blockDim.x + threadIdx.x;
  if (idx >= NP0) return;
  int x = idx % W0; int t = idx / W0; int y = t % H0; int z = t / H0;
  float acc = bias[0];
  for (int ci = 0; ci < 16; ++ci)
    for (int kz = 0; kz < 3; ++kz) {
      int zi = z - 1 + kz; if ((unsigned)zi >= (unsigned)D0) continue;
      for (int ky = 0; ky < 3; ++ky) {
        int yi = y - 1 + ky; if ((unsigned)yi >= (unsigned)H0) continue;
        for (int kx = 0; kx < 3; ++kx) {
          int xi = x - 1 + kx; if ((unsigned)xi >= (unsigned)W0) continue;
          acc += (float)in[(size_t)ci*NP0 + ((size_t)zi*H0 + yi)*W0 + xi]
               * wt[ci*27 + kz*9 + ky*3 + kx];
        }
      }
    }
  out[idx] = acc;
}

// ----------------------------------- softmax over D, depth, 4-tap confidence
__global__ void mvs_softmax_depth(const float* __restrict__ creg,
                                  const float* __restrict__ hypo,
                                  float* __restrict__ depth_out,
                                  float* __restrict__ conf_out,
                                  float* __restrict__ pv_out) {
  int pix = blockIdx.x*blockDim.x + threadIdx.x;
  if (pix >= HW0) return;
  float p[D0];
  float mx = -1e30f;
  for (int d = 0; d < D0; ++d) { float v = creg[(size_t)d*HW0 + pix]; p[d] = v; if (v > mx) mx = v; }
  float s = 0.f;
  for (int d = 0; d < D0; ++d) { float e = __expf(p[d]-mx); p[d] = e; s += e; }
  float is = 1.f / s;
  float dep = 0.f, didx = 0.f;
  for (int d = 0; d < D0; ++d) {
    p[d] *= is;
    pv_out[(size_t)d*HW0 + pix] = p[d];
    dep  += p[d] * hypo[(size_t)d*HW0 + pix];
    didx += p[d] * (float)d;
  }
  depth_out[pix] = dep;
  int di = (int)didx;
  if (di < 0) di = 0;
  if (di > D0-1) di = D0-1;
  float c4 = 0.f;
  for (int u = di-1; u <= di+2; ++u) if (u >= 0 && u < D0) c4 += p[u];
  conf_out[pix] = c4;
}

// ============================================================================
extern "C" void kernel_launch(void* const* d_in, const int* in_sizes, int n_in,
                              void* d_out, int out_size, void* d_ws, size_t ws_size,
                              hipStream_t stream) {
  const float* ref_feature  = (const float*)d_in[0];
  const float* src_features = (const float*)d_in[1];
  const float* ref_proj     = (const float*)d_in[2];
  const float* src_projs    = (const float*)d_in[3];
  const float* depth_hypos  = (const float*)d_in[4];
  const float* Wl[10]; const float* Gl[10]; const float* Bl[10];
  int pi = 5;
  for (int i = 0; i < 10; ++i) {
    Wl[i] = (const float*)d_in[pi++];
    Gl[i] = (const float*)d_in[pi++];
    Bl[i] = (const float*)d_in[pi++];
  }
  const float* prob_w = (const float*)d_in[pi++];
  const float* prob_b = (const float*)d_in[pi++];

  char* wsb = (char*)d_ws;
  size_t off = 0;
  auto alloc = [&](size_t bytes) { char* p = wsb + off; off += (bytes + 255) & ~(size_t)255; return p; };
  float*     projbuf = (float*)alloc(32*sizeof(float));
  float*     stats   = (float*)alloc(128*sizeof(float));
  float*     convout = (float*)alloc((size_t)16*NP0*sizeof(float));  // shared f32 conv scratch
  _Float16*  A16     = (_Float16*)alloc((size_t)16*NP0*2);           // full-res 16ch
  _Float16*  B16     = (_Float16*)alloc((size_t)16*NP0*2);
  _Float16*  C16     = (_Float16*)alloc((size_t)64*NP1*2);           // half-res up to 64ch
  _Float16*  D16     = (_Float16*)alloc((size_t)64*NP1*2);
  _Float16*  E16     = (_Float16*)alloc((size_t)64*NP1*2);
  float*     creg    = (float*)alloc((size_t)NP0*sizeof(float));

  float* depth_out = (float*)d_out;
  float* conf_out  = depth_out + HW0;
  float* pv_out    = conf_out + HW0;

  mvs_proj_kernel<<<1, 1, 0, stream>>>(ref_proj, src_projs, projbuf);
  mvs_cost_volume<<<(NP0+255)/256, 256, 0, stream>>>(ref_feature, src_features,
                                                     depth_hypos, projbuf, A16);

  auto conv = [&](const _Float16* in, const float* w, float* out, int Cin, int Cout,
                  int Din,int Hin,int Win, int Dout,int Hout,int Wout, int stride) {
    int npo = Dout*Hout*Wout;
    dim3 grid((npo + 127)/128, Cout/16);   // 4 waves x 32 rows per block
    mvs_conv3d_wmma<<<grid, 128, 0, stream>>>(in, w, out, Cin,
                                              Din,Hin,Win, Dout,Hout,Wout, stride);
  };
  auto bnrelu = [&](int li, const _Float16* skip, _Float16* y, int C, int npix) {
    mvs_bn_stats<<<C, 256, 0, stream>>>(convout, stats, npix);
    dim3 grid((npix+255)/256, C);
    mvs_bn_relu_add<<<grid, 256, 0, stream>>>(convout, stats, Gl[li], Bl[li],
                                              skip, y, npix);
  };

  // encoder
  conv(A16, Wl[0], convout, 16, 16, D0,H0,W0, D0,H0,W0, 1);  bnrelu(0, nullptr, B16, 16, NP0);
  conv(B16, Wl[1], convout, 16, 16, D0,H0,W0, D0,H0,W0, 1);  bnrelu(1, nullptr, A16, 16, NP0);  // a1 skip
  conv(A16, Wl[2], convout, 16, 32, D0,H0,W0, D1,H1,W1, 2);  bnrelu(2, nullptr, C16, 32, NP1);
  conv(C16, Wl[3], convout, 32, 32, D1,H1,W1, D1,H1,W1, 1);  bnrelu(3, nullptr, D16, 32, NP1);
  conv(D16, Wl[4], convout, 32, 32, D1,H1,W1, D1,H1,W1, 1);  bnrelu(4, nullptr, C16, 32, NP1);  // a4 skip
  conv(C16, Wl[5], convout, 32, 64, D1,H1,W1, D1,H1,W1, 1);  bnrelu(5, nullptr, D16, 64, NP1);
  conv(D16, Wl[6], convout, 64, 64, D1,H1,W1, D1,H1,W1, 1);  bnrelu(6, nullptr, E16, 64, NP1);
  conv(E16, Wl[7], convout, 64, 64, D1,H1,W1, D1,H1,W1, 1);  bnrelu(7, nullptr, D16, 64, NP1);
  // decoder
  conv(D16, Wl[8], convout, 64, 32, D1,H1,W1, D1,H1,W1, 1);  bnrelu(8, C16, E16, 32, NP1);      // c8 = a4 + bnrelu
  {
    int total = 16*NP0;
    mvs_convT_s2<<<(total+255)/256, 256, 0, stream>>>(E16, Wl[9], convout);
  }
  bnrelu(9, A16, B16, 16, NP0);                                                                 // c9 = a1 + bnrelu
  mvs_prob_conv<<<(NP0+255)/256, 256, 0, stream>>>(B16, prob_w, prob_b, creg);
  mvs_softmax_depth<<<(HW0+127)/128, 128, 0, stream>>>(creg, depth_hypos,
                                                       depth_out, conf_out, pv_out);

  (void)in_sizes; (void)n_in; (void)out_size; (void)ws_size;
}
